// MQA_34059090657408
// MI455X (gfx1250) — matrix operand order repaired and checked by reading, not compiled, not run on hardware
//
#include <hip/hip_runtime.h>

// Problem constants (match reference)
constexpr int Bc  = 4;
constexpr int Tc  = 2048;
constexpr int Dc  = 1024;
constexpr int Hc  = 16;
constexpr int HDc = 64;

typedef _Float16 v16h __attribute__((ext_vector_type(16)));
typedef float    v8f  __attribute__((ext_vector_type(8)));

#define WMMA_F32_F16(a, b, c) \
  __builtin_amdgcn_wmma_f32_16x16x32_f16(false, (a), false, (b), (short)0, (c), false, false)

// ---------------------------------------------------------------------------
// CDNA5 async global->LDS copy (CDNA5 ISA), ASYNCcnt-tracked.
// ---------------------------------------------------------------------------
__device__ inline unsigned lds_off_u32(const void* p) {
  return (unsigned)(uintptr_t)(__attribute__((address_space(3))) const void*)p;
}
__device__ inline void async_copy_b128(unsigned lds_off, const void* gaddr) {
  asm volatile("global_load_async_to_lds_b128 %0, %1, off"
               :: "v"(lds_off), "v"(gaddr)
               : "memory");
}
__device__ inline void wait_async_all() {
  asm volatile("s_wait_asynccnt 0x0" ::: "memory");
}

// ---------------------------------------------------------------------------
// Fragment loaders (wave32 CDNA5 WMMA layouts, ISA 7.12.2); both read two
// 8-half runs per lane (k = 8h.., 16+8h..) -> 2x ds_load_b128 each.
// ---------------------------------------------------------------------------
__device__ inline v16h load_a_16x32(const _Float16* base, int ld) {
  const int lane = threadIdx.x & 31;
  const int k0 = (lane & 16) ? 8 : 0;
  const _Float16* p = base + (lane & 15) * ld + k0;
  v16h a;
#pragma unroll
  for (int i = 0; i < 8; ++i) { a[i] = p[i]; a[8 + i] = p[16 + i]; }
  return a;
}

__device__ inline v16h load_b_klast(const _Float16* base, int ld) {
  const int lane = threadIdx.x & 31;
  const int k0 = (lane & 16) ? 8 : 0;
  const _Float16* p = base + (lane & 15) * ld + k0;
  v16h b;
#pragma unroll
  for (int i = 0; i < 8; ++i) { b[i] = p[i]; b[8 + i] = p[16 + i]; }
  return b;
}

// ---------------------------------------------------------------------------
__global__ void f32_to_f16_kernel(const float* __restrict__ in,
                                  _Float16* __restrict__ out, int n) {
  int i = blockIdx.x * blockDim.x + threadIdx.x;
  int stride = gridDim.x * blockDim.x;
  for (; i < n; i += stride) out[i] = (_Float16)in[i];
}

// v [B*T, 64] -> vT [B, 64, T]
__global__ void transpose_v_kernel(const _Float16* __restrict__ v,
                                   _Float16* __restrict__ vT) {
  int i = blockIdx.x * blockDim.x + threadIdx.x;  // < B*T*64
  int hd = i & 63;
  int bt = i >> 6;
  int b = bt >> 11;  // T = 2048
  int t = bt & (Tc - 1);
  vT[((size_t)b * HDc + hd) * Tc + t] = v[i];
}

// fp32 suffix sums of V at 32-key granularity:
// suf[b][kc][hd] = sum_{t >= kc*32} v[b,t,hd]
__global__ void v_suffix_kernel(const _Float16* __restrict__ v,
                                float* __restrict__ suf) {
  int b = blockIdx.x;
  int hd = threadIdx.x;  // 64 threads
  float acc = 0.0f;
  for (int kc = Tc / 32 - 1; kc >= 0; --kc) {
    for (int t = 31; t >= 0; --t)
      acc += (float)v[((size_t)b * Tc + kc * 32 + t) * HDc + hd];
    suf[((size_t)b * (Tc / 32) + kc) * HDc + hd] = acc;
  }
}

// ---------------------------------------------------------------------------
// Tiled WMMA GEMM: C[M,N] = A[M,K] @ B[K,N], f16 in, f32 accumulate.
// 256 threads = 8 waves (4x2); block tile 128 x BN; wave tile 32 x BN/2.
// Double-buffered: A via async b128 copies, B via vector global load +
// transposed LDS stores ([N][K]) so fragments are contiguous ds_load_b128.
// ---------------------------------------------------------------------------
template <int BN, bool F32OUT>
__global__ __launch_bounds__(256) void gemm_f16_kernel(
    const _Float16* __restrict__ A, const _Float16* __restrict__ Bm,
    void* __restrict__ Cout, int M, int N, int K) {
  constexpr int NFRAG = BN / 32;
  __shared__ __align__(16) _Float16 Alds[2][128 * 32];
  __shared__ __align__(16) _Float16 BldsT[2][BN * 32];

  const int tid = threadIdx.x;
  const int wave = tid >> 5;
  const int lane = tid & 31;
  const int lanecol = lane & 15;
  const int upper = (lane >> 4) & 1;
  const int wm = wave & 3;
  const int wn = wave >> 2;
  const int M0 = blockIdx.y * 128;
  const int N0 = blockIdx.x * BN;

  auto stageA = [&](int k0, int buf) {
#pragma unroll
    for (int t = 0; t < 2; ++t) {
      int idx8 = (tid + t * 256) * 8;
      int r = idx8 >> 5, c = idx8 & 31;
      async_copy_b128(lds_off_u32(&Alds[buf][idx8]),
                      A + (size_t)(M0 + r) * K + (k0 + c));
    }
  };
  int4 breg[BN / 64];
  auto loadB = [&](int k0) {
#pragma unroll
    for (int t = 0; t < BN / 64; ++t) {
      int idx8 = (tid + t * 256) * 8;
      int r = idx8 / BN, c = idx8 & (BN - 1);
      breg[t] =
          *reinterpret_cast<const int4*>(Bm + (size_t)(k0 + r) * N + (N0 + c));
    }
  };
  auto storeB = [&](int buf) {
#pragma unroll
    for (int t = 0; t < BN / 64; ++t) {
      int idx8 = (tid + t * 256) * 8;
      int r = idx8 / BN, c = idx8 & (BN - 1);
      const _Float16* h = reinterpret_cast<const _Float16*>(&breg[t]);
#pragma unroll
      for (int j = 0; j < 8; ++j) BldsT[buf][(c + j) * 32 + r] = h[j];
    }
  };

  v8f acc[2][NFRAG];
#pragma unroll
  for (int mi = 0; mi < 2; ++mi)
#pragma unroll
    for (int ni = 0; ni < NFRAG; ++ni) acc[mi][ni] = v8f{};

  stageA(0, 0);
  loadB(0);
  storeB(0);
  wait_async_all();
  __syncthreads();

  for (int k0 = 0; k0 < K; k0 += 32) {
    const int cur = (k0 >> 5) & 1;
    const int nk = (k0 + 32 < K) ? k0 + 32 : 0;  // clamp: harmless re-read
    stageA(nk, cur ^ 1);
    loadB(nk);

    v16h af[2], bf[NFRAG];
#pragma unroll
    for (int mi = 0; mi < 2; ++mi)
      af[mi] = load_a_16x32(&Alds[cur][(wm * 32 + mi * 16) * 32], 32);
#pragma unroll
    for (int ni = 0; ni < NFRAG; ++ni)
      bf[ni] = load_b_klast(&BldsT[cur][(wn * (BN / 2) + ni * 16) * 32], 32);
#pragma unroll
    for (int mi = 0; mi < 2; ++mi)
#pragma unroll
      for (int ni = 0; ni < NFRAG; ++ni)
        acc[mi][ni] = WMMA_F32_F16(af[mi], bf[ni], acc[mi][ni]);

    storeB(cur ^ 1);
    wait_async_all();
    __syncthreads();
  }

#pragma unroll
  for (int mi = 0; mi < 2; ++mi)
#pragma unroll
    for (int ni = 0; ni < NFRAG; ++ni)
#pragma unroll
      for (int r = 0; r < 8; ++r) {
        int row = M0 + wm * 32 + mi * 16 + upper * 8 + r;
        int col = N0 + wn * (BN / 2) + ni * 16 + lanecol;
        if (F32OUT)
          ((float*)Cout)[(size_t)row * N + col] = acc[mi][ni][r];
        else
          ((_Float16*)Cout)[(size_t)row * N + col] = (_Float16)acc[mi][ni][r];
      }
}

// ---------------------------------------------------------------------------
// Flash-style MQA attention, faithful to masked_fill(-1e-9).
// Streams only the causal band of key chunks; the fully-masked tail (score
// == -1e-9 for every key) is folded in closed form using precomputed V
// suffix sums: l += ntail*p, O += p*suffix, p = exp(-1e-9 - m).
// Computes S^T = K @ Q^T so the C-layout of S^T is exactly the A-fragment
// layout of P. 128 threads = 4 waves; wave owns 16 queries.
// ---------------------------------------------------------------------------
__global__ __launch_bounds__(128) void mqa_attn_kernel(
    const _Float16* __restrict__ q, const _Float16* __restrict__ k,
    const _Float16* __restrict__ vT, const float* __restrict__ suf,
    _Float16* __restrict__ out) {
  __shared__ __align__(16) _Float16 Klds[2][32 * 64];   // [key][hd]
  __shared__ __align__(16) _Float16 VldsT[2][64 * 32];  // [hd][key]

  const int tid = threadIdx.x;
  const int wave = tid >> 5;
  const int lane = tid & 31;
  const int lanecol = lane & 15;
  const int upper = (lane >> 4) & 1;
  const int b = blockIdx.z;
  const int h = blockIdx.y;
  const int qt = blockIdx.x;
  const int qrow0 = qt * 64 + wave * 16;
  const float scale = 0.125f;  // 1/sqrt(64)

  // chunks containing any key <= last query of this 64-row tile (uniform
  // across the block so cooperative staging barriers stay legal)
  const int KC_block = qt * 2 + 2;
  const int ntail = Tc - KC_block * 32;

  auto stage = [&](int kc, int buf) {
#pragma unroll
    for (int t = 0; t < 2; ++t) {
      int idx8 = (tid + t * 128) * 8;
      {
        int r = idx8 >> 6, c = idx8 & 63;
        async_copy_b128(lds_off_u32(&Klds[buf][idx8]),
                        k + (size_t)(b * Tc + kc * 32 + r) * HDc + c);
      }
      {
        int r = idx8 >> 5, c = idx8 & 31;
        async_copy_b128(lds_off_u32(&VldsT[buf][idx8]),
                        vT + ((size_t)b * HDc + r) * Tc + (kc * 32 + c));
      }
    }
  };

  // Q^T as B-fragments (contraction = hd, col = query)
  v16h qb[2];
  {
    const _Float16* qp =
        q + (size_t)(b * Tc + qrow0 + lanecol) * Dc + h * HDc + upper * 8;
#pragma unroll
    for (int c = 0; c < 2; ++c) {
      const _Float16* p = qp + c * 32;
#pragma unroll
      for (int i = 0; i < 8; ++i) { qb[c][i] = p[i]; qb[c][8 + i] = p[16 + i]; }
    }
  }

  v8f oacc[4];
#pragma unroll
  for (int nt = 0; nt < 4; ++nt) oacc[nt] = v8f{};
  float m_l = -__builtin_inff();  // per-lane: running max for query=lanecol
  float l_l = 0.0f;               // per-lane: running sum for query=lanecol

  stage(0, 0);
  wait_async_all();
  __syncthreads();

  for (int kc = 0; kc < KC_block; ++kc) {
    const int cur = kc & 1;
    const int nxt = (kc + 1 < KC_block) ? kc + 1 : 0;  // clamp
    stage(nxt, cur ^ 1);

    // S^T tiles: c0 = keys 0-15 (local), c1 = keys 16-31; contraction hd=64
    v8f c0 = v8f{}, c1 = v8f{};
    c0 = WMMA_F32_F16(load_a_16x32(&Klds[cur][0 * 16 * 64 + 0], 64), qb[0], c0);
    c0 = WMMA_F32_F16(load_a_16x32(&Klds[cur][0 * 16 * 64 + 32], 64), qb[1], c0);
    c1 = WMMA_F32_F16(load_a_16x32(&Klds[cur][1 * 16 * 64 + 0], 64), qb[0], c1);
    c1 = WMMA_F32_F16(load_a_16x32(&Klds[cur][1 * 16 * 64 + 32], 64), qb[1], c1);

    const int qg = qrow0 + lanecol;
    float a0[8], a1[8];
    if (kc * 32 + 31 <= qrow0) {  // wave-uniform: fully unmasked chunk
#pragma unroll
      for (int r = 0; r < 8; ++r) {
        a0[r] = c0[r] * scale;
        a1[r] = c1[r] * scale;
      }
    } else {  // diagonal / masked chunk for this wave
#pragma unroll
      for (int r = 0; r < 8; ++r) {
        int kg = kc * 32 + upper * 8 + r;
        a0[r] = (kg > qg) ? -1e-9f : c0[r] * scale;
        a1[r] = (kg + 16 > qg) ? -1e-9f : c1[r] * scale;
      }
    }

    // per-lane chunk max (16 values) + one xor-16 combine
    float mx = fmaxf(a0[0], a1[0]);
#pragma unroll
    for (int r = 1; r < 8; ++r) mx = fmaxf(mx, fmaxf(a0[r], a1[r]));
    mx = fmaxf(mx, __shfl_xor(mx, 16, 32));
    float mn = fmaxf(m_l, mx);
    float f = __expf(m_l - mn);  // first chunk: exp(-inf)=0
    m_l = mn;

    // P in registers: C-layout of S^T == A-fragment layout of P
    v16h pf;
    float s = 0.0f;
#pragma unroll
    for (int r = 0; r < 8; ++r) {
      float e0 = __expf(a0[r] - mn);
      float e1 = __expf(a1[r] - mn);
      s += e0 + e1;
      pf[r] = (_Float16)e0;
      pf[8 + r] = (_Float16)e1;
    }
    s += __shfl_xor(s, 16, 32);
    l_l = l_l * f + s;

    // rescale O: row (query) upper*8+r needs f from lane upper*8+r
#pragma unroll
    for (int r = 0; r < 8; ++r) {
      float fr = __shfl(f, upper * 8 + r, 32);
#pragma unroll
      for (int nt = 0; nt < 4; ++nt) oacc[nt][r] *= fr;
    }

    // O += P @ V (K-depth 32 keys, N = 64 hd as four 16-wide tiles)
#pragma unroll
    for (int nt = 0; nt < 4; ++nt)
      oacc[nt] =
          WMMA_F32_F16(pf, load_b_klast(&VldsT[cur][nt * 16 * 32], 32), oacc[nt]);

    wait_async_all();
    __syncthreads();
  }

  // Closed-form masked tail: every tail key scores exactly -1e-9.
  if (ntail > 0) {
    float m2 = fmaxf(m_l, -1e-9f);
    float fadj = __expf(m_l - m2);       // rescale past contributions
    float pt = __expf(-1e-9f - m2);      // weight of each tail key
    l_l = l_l * fadj + (float)ntail * pt;
    float sv[4];
#pragma unroll
    for (int nt = 0; nt < 4; ++nt)
      sv[nt] = suf[((size_t)b * (Tc / 32) + KC_block) * HDc + nt * 16 + lanecol];
#pragma unroll
    for (int r = 0; r < 8; ++r) {
      float fr = __shfl(fadj, upper * 8 + r, 32);
      float pr = __shfl(pt, upper * 8 + r, 32);
#pragma unroll
      for (int nt = 0; nt < 4; ++nt)
        oacc[nt][r] = oacc[nt][r] * fr + pr * sv[nt];
    }
  }

  // Epilogue: normalize (l per row via one-time broadcast) and store f16
#pragma unroll
  for (int r = 0; r < 8; ++r) {
    float lr = __shfl(l_l, upper * 8 + r, 32);
    float inv = 1.0f / lr;
    int row = b * Tc + qrow0 + upper * 8 + r;
#pragma unroll
    for (int nt = 0; nt < 4; ++nt) {
      int col = h * HDc + nt * 16 + lanecol;
      out[(size_t)row * Dc + col] = (_Float16)(oacc[nt][r] * inv);
    }
  }
}

// ---------------------------------------------------------------------------
// Launch
// ---------------------------------------------------------------------------
extern "C" void kernel_launch(void* const* d_in, const int* in_sizes, int n_in,
                              void* d_out, int out_size, void* d_ws,
                              size_t ws_size, hipStream_t stream) {
  (void)in_sizes; (void)n_in; (void)out_size; (void)ws_size;
  const float* x  = (const float*)d_in[0];
  // d_in[1] is the mask: deterministic tril(T,T) -> computed analytically
  const float* wq = (const float*)d_in[2];
  const float* wk = (const float*)d_in[3];
  const float* wv = (const float*)d_in[4];
  const float* wo = (const float*)d_in[5];
  float* out = (float*)d_out;

  char* w = (char*)d_ws;
  auto take = [&](size_t bytes) { void* p = (void*)w; w += bytes; return p; };
  _Float16* xh  = (_Float16*)take((size_t)Bc * Tc * Dc * 2);
  _Float16* wqh = (_Float16*)take((size_t)Dc * Dc * 2);
  _Float16* wkh = (_Float16*)take((size_t)Dc * HDc * 2);
  _Float16* wvh = (_Float16*)take((size_t)Dc * HDc * 2);
  _Float16* woh = (_Float16*)take((size_t)Dc * Dc * 2);
  _Float16* qh  = (_Float16*)take((size_t)Bc * Tc * Dc * 2);
  _Float16* kh  = (_Float16*)take((size_t)Bc * Tc * HDc * 2);
  _Float16* vh  = (_Float16*)take((size_t)Bc * Tc * HDc * 2);
  _Float16* vth = (_Float16*)take((size_t)Bc * HDc * Tc * 2);
  _Float16* ah  = (_Float16*)take((size_t)Bc * Tc * Dc * 2);
  float*    sufp= (float*)take((size_t)Bc * (Tc / 32) * HDc * 4);

  f32_to_f16_kernel<<<1024, 256, 0, stream>>>(x, xh, Bc * Tc * Dc);
  f32_to_f16_kernel<<<512, 256, 0, stream>>>(wq, wqh, Dc * Dc);
  f32_to_f16_kernel<<<64, 256, 0, stream>>>(wk, wkh, Dc * HDc);
  f32_to_f16_kernel<<<64, 256, 0, stream>>>(wv, wvh, Dc * HDc);
  f32_to_f16_kernel<<<512, 256, 0, stream>>>(wo, woh, Dc * Dc);

  const int MT = Bc * Tc;  // 8192
  {  // q = x @ wq
    dim3 g(Dc / 128, MT / 128);
    gemm_f16_kernel<128, false><<<g, 256, 0, stream>>>(xh, wqh, qh, MT, Dc, Dc);
  }
  {  // k,v = x @ wk / wv
    dim3 g(HDc / 64, MT / 128);
    gemm_f16_kernel<64, false><<<g, 256, 0, stream>>>(xh, wkh, kh, MT, HDc, Dc);
    gemm_f16_kernel<64, false><<<g, 256, 0, stream>>>(xh, wvh, vh, MT, HDc, Dc);
  }
  transpose_v_kernel<<<(Bc * Tc * HDc) / 256, 256, 0, stream>>>(vh, vth);
  v_suffix_kernel<<<Bc, HDc, 0, stream>>>(vh, sufp);
  {  // attention
    dim3 g(Tc / 64, Hc, Bc);
    mqa_attn_kernel<<<g, 128, 0, stream>>>(qh, kh, vth, sufp, ah);
  }
  {  // out = attn @ wo -> fp32
    dim3 g(Dc / 128, MT / 128);
    gemm_f16_kernel<128, true><<<g, 256, 0, stream>>>(ah, woh, out, MT, Dc, Dc);
  }
}
